// GIN_36687610642606
// MI455X (gfx1250) — matrix operand order repaired and checked by reading, not compiled, not run on hardware
//
#include <hip/hip_runtime.h>
#include <hip/hip_bf16.h>
#include <math.h>

// ---------------------------------------------------------------------------
// GIN forward for MI455X (gfx1250, wave32).
// GEMMs via V_WMMA_F32_16X16X32_BF16, N-blocked 4 tiles/wave so one A fragment
// feeds 4 WMMAs. B fragments are loaded as a batch per k-step so the backend
// can stagger s_wait_loadcnt across the WMMA quartet instead of draining.
// SpMM via edge-parallel global_atomic_add_f32 (x fits in 192MB L2).
// ---------------------------------------------------------------------------

#define GNN_N     100000
#define GNN_E     3200000
#define GNN_K     256       // NFEAT == NHID == 256
#define GNN_C     40        // NCLASS
#define GNN_CPAD  48        // NCLASS padded to 3 N-tiles of 16

typedef __attribute__((ext_vector_type(16))) __bf16 v16bf;
typedef __attribute__((ext_vector_type(8)))  __bf16 v8bf;
typedef __attribute__((ext_vector_type(8)))  float  v8f;

union AFrag { v16bf v; v8bf h[2]; };

// ---- elementwise helpers ---------------------------------------------------

__global__ void copy_f4_kernel(const float4* __restrict__ src,
                               float4* __restrict__ dst, long long n4) {
    long long i = (long long)blockIdx.x * blockDim.x + threadIdx.x;
    if (i < n4) dst[i] = src[i];
}

__global__ void f32_to_bf16_kernel(const float* __restrict__ src,
                                   __bf16* __restrict__ dst, long long n) {
    long long i = (long long)blockIdx.x * blockDim.x + threadIdx.x;
    if (i < n) dst[i] = (__bf16)src[i];
}

// Wt[n*K + k] = bf16(W[k*Nout + n]); zero-pad rows n >= Nout up to Npad.
__global__ void weight_transpose_kernel(const float* __restrict__ W,
                                        __bf16* __restrict__ Wt,
                                        int K, int Nout, int Npad) {
    int i = blockIdx.x * blockDim.x + threadIdx.x;
    if (i >= Npad * K) return;
    int n = i / K;
    int k = i - n * K;
    float v = (n < Nout) ? W[k * Nout + n] : 0.0f;
    Wt[i] = (__bf16)v;
}

// ---- SpMM: acc[row] += w * src[col], edge-parallel, 4 feats per thread -----

__global__ void edge_scatter_kernel(const int* __restrict__ erow,
                                    const int* __restrict__ ecol,
                                    const float* __restrict__ ew,
                                    const float* __restrict__ src,
                                    float* __restrict__ acc) {
    long long t = (long long)blockIdx.x * blockDim.x + threadIdx.x;
    long long e = t >> 6;          // 64 threads per edge (256 feats / 4)
    if (e >= GNN_E) return;
    int fg = (int)(t & 63);
    int r = erow[e];
    int c = ecol[e];
    float w = ew[e];
    const float4 xv = *(const float4*)(src + (size_t)c * GNN_K + fg * 4);
    float* dst = acc + (size_t)r * GNN_K + fg * 4;
    atomicAdd(dst + 0, w * xv.x);
    atomicAdd(dst + 1, w * xv.y);
    atomicAdd(dst + 2, w * xv.z);
    atomicAdd(dst + 3, w * xv.w);
}

// ---- WMMA GEMM: one wave computes a 16 x (16*NB) strip; A frag reused ------
// A: [M x 256] bf16 row-major.  Wt: [Npad x 256] bf16 (pre-transposed).
// Bias folded into accumulator init (C/D element r <-> row, lane <-> col).

template <int NB>
__global__ void wmma_gemm_kernel(const __bf16* __restrict__ A,
                                 const __bf16* __restrict__ Wt,
                                 const float* __restrict__ bias,
                                 __bf16* __restrict__ outbf,   // may be null
                                 float*  __restrict__ outf,    // may be null
                                 int M, int Npad, int Nvalid, int ldout,
                                 int do_relu) {
    const int wavesPerBlock = blockDim.x >> 5;
    const int wid   = blockIdx.x * wavesPerBlock + (threadIdx.x >> 5);
    const int ngrp  = (Npad >> 4) / NB;               // N-tile groups per row
    const int total = (M >> 4) * ngrp;
    if (wid >= total) return;                 // wave-uniform: EXEC stays all-1s

    const int tm   = wid / ngrp;
    const int tg   = wid - tm * ngrp;                 // N-tile group index
    const int lane = threadIdx.x & 31;
    const int hlf  = lane >> 4;               // 0: lanes 0-15, 1: lanes 16-31
    const int l15  = lane & 15;

    // A: lane holds row (tm*16 + l15); K=16*hlf+e.
    const __bf16* arow = A + (size_t)(tm * 16 + l15) * GNN_K;

    // B: per sub-tile j, lane holds col ((tg*NB+j)*16 + l15); K=16*hlf+e.
    const __bf16* brow[NB];
    int   ncol[NB];
    v8f   c[NB];
#pragma unroll
    for (int j = 0; j < NB; ++j) {
        ncol[j] = (tg * NB + j) * 16 + l15;
        brow[j] = Wt + (size_t)ncol[j] * GNN_K;
        float bv = (ncol[j] < Nvalid) ? bias[ncol[j]] : 0.0f;
        v8f ci = {bv, bv, bv, bv, bv, bv, bv, bv};
        c[j] = ci;                       // bias pre-loaded into accumulator
    }

#pragma unroll
    for (int k0 = 0; k0 < GNN_K; k0 += 32) {
        // Issue all loads for this k-step first: 2x A (16B) + NB x B (32B)
        // live simultaneously so waits can stagger across the WMMA chain.
        AFrag a;
        a.h[0] = *(const v8bf*)(arow + k0 + 16 * hlf);
        a.h[1] = *(const v8bf*)(arow + k0 + 16 * hlf + 8);
        v16bf b[NB];
#pragma unroll
        for (int j = 0; j < NB; ++j)
            b[j] = *(const v16bf*)(brow[j] + k0 + 16 * hlf);
#pragma unroll
        for (int j = 0; j < NB; ++j)
            c[j] = __builtin_amdgcn_wmma_f32_16x16x32_bf16(
                       /*neg_a=*/false, a.v, /*neg_b=*/false, b[j],
                       /*c_mod=*/(short)0, c[j],
                       /*reuse_a=*/false, /*reuse_b=*/false);
    }

    // Epilogue: element r -> row tm*16 + 8*hlf + r, col ncol[j]
    const int rowbase = tm * 16 + 8 * hlf;
#pragma unroll
    for (int j = 0; j < NB; ++j) {
        const int n = ncol[j];
#pragma unroll
        for (int r = 0; r < 8; ++r) {
            float v = c[j][r];
            if (do_relu) v = fmaxf(v, 0.0f);
            const size_t row = (size_t)(rowbase + r);
            if (outbf)               outbf[row * Npad + n]         = (__bf16)v;
            if (outf && n < Nvalid)  outf[row * (size_t)ldout + n] = v;
        }
    }
}

// ---- log_softmax over 40 classes, in place ---------------------------------

__global__ void log_softmax_kernel(float* __restrict__ out, int M, int C) {
    int i = blockIdx.x * blockDim.x + threadIdx.x;
    if (i >= M) return;
    float* r = out + (size_t)i * C;
    float mx = r[0];
    for (int c = 1; c < C; ++c) mx = fmaxf(mx, r[c]);
    float s = 0.0f;
    for (int c = 0; c < C; ++c) s += expf(r[c] - mx);
    float ls = logf(s) + mx;
    for (int c = 0; c < C; ++c) r[c] = r[c] - ls;
}

// ---------------------------------------------------------------------------

extern "C" void kernel_launch(void* const* d_in, const int* in_sizes, int n_in,
                              void* d_out, int out_size, void* d_ws, size_t ws_size,
                              hipStream_t stream) {
    const float* x    = (const float*)d_in[0];
    const int*   erow = (const int*)  d_in[1];
    const int*   ecol = (const int*)  d_in[2];
    const float* ew   = (const float*)d_in[3];
    const float* W1 = (const float*)d_in[4];  const float* b1 = (const float*)d_in[5];
    const float* W2 = (const float*)d_in[6];  const float* b2 = (const float*)d_in[7];
    const float* W3 = (const float*)d_in[8];  const float* b3 = (const float*)d_in[9];
    const float* W4 = (const float*)d_in[10]; const float* b4 = (const float*)d_in[11];

    // ---- workspace layout (~308 MB) ----
    char* ws = (char*)d_ws;
    const size_t SZ_F32 = (size_t)GNN_N * GNN_K * 4;   // 102,400,000
    const size_t SZ_BF  = (size_t)GNN_N * GNN_K * 2;   //  51,200,000
    float*  accA = (float*)(ws);                       // f32 activations / h2
    float*  accB = (float*)(ws + SZ_F32);              // spmm accumulator
    __bf16* hbA  = (__bf16*)(ws + 2 * SZ_F32);         // bf16 activations ping
    __bf16* hbB  = (__bf16*)(ws + 2 * SZ_F32 + SZ_BF); // bf16 activations pong
    __bf16* w1t  = (__bf16*)(ws + 2 * SZ_F32 + 2 * SZ_BF);
    __bf16* w2t  = w1t + (size_t)GNN_K * GNN_K;
    __bf16* w3t  = w2t + (size_t)GNN_K * GNN_K;
    __bf16* w4t  = w3t + (size_t)GNN_K * GNN_K;        // padded 48 x 256

    const long long nElem = (long long)GNN_N * GNN_K;  // 25.6M
    const long long n4    = nElem / 4;

    // ---- weights -> transposed bf16 ----
    {
        int e = GNN_K * GNN_K;
        weight_transpose_kernel<<<(e + 255) / 256, 256, 0, stream>>>(W1, w1t, GNN_K, GNN_K, GNN_K);
        weight_transpose_kernel<<<(e + 255) / 256, 256, 0, stream>>>(W2, w2t, GNN_K, GNN_K, GNN_K);
        weight_transpose_kernel<<<(e + 255) / 256, 256, 0, stream>>>(W3, w3t, GNN_K, GNN_K, GNN_K);
        int e4 = GNN_CPAD * GNN_K;
        weight_transpose_kernel<<<(e4 + 255) / 256, 256, 0, stream>>>(W4, w4t, GNN_K, GNN_C, GNN_CPAD);
    }

    const long long scatterThreads = (long long)GNN_E * 64;
    const int scatterBlocks = (int)((scatterThreads + 255) / 256);
    const int elemBlocks4 = (int)((n4 + 255) / 256);
    const int elemBlocks1 = (int)((nElem + 255) / 256);

    // ---- h0 = x + spmm(x) ----
    copy_f4_kernel<<<elemBlocks4, 256, 0, stream>>>((const float4*)x, (float4*)accA, n4);
    edge_scatter_kernel<<<scatterBlocks, 256, 0, stream>>>(erow, ecol, ew, x, accA);
    f32_to_bf16_kernel<<<elemBlocks1, 256, 0, stream>>>(accA, hbA, nElem);

    // GEMM tiling: 16x64 strip per wave (NB=4), 8 waves (256 thr) per block.
    const int waves256 = (GNN_N / 16) * ((GNN_K / 16) / 4);     // 25000
    const int gemmBlocks256 = (waves256 + 7) / 8;               // 3125
    const int wavesOut = (GNN_N / 16) * ((GNN_CPAD / 16) / 3);  // 6250
    const int gemmBlocksOut = (wavesOut + 7) / 8;

    // ---- h1 = relu(h0 @ W1 + b1)  (bf16 out) ----
    wmma_gemm_kernel<4><<<gemmBlocks256, 256, 0, stream>>>(
        hbA, w1t, b1, hbB, nullptr, GNN_N, GNN_K, GNN_K, GNN_K, 1);

    // ---- h2 = relu(h1 @ W2 + b2)  (f32 out, needed by spmm) ----
    wmma_gemm_kernel<4><<<gemmBlocks256, 256, 0, stream>>>(
        hbB, w2t, b2, nullptr, accA, GNN_N, GNN_K, GNN_K, GNN_K, 1);

    // ---- h3 = h2 + spmm(h2) ----
    copy_f4_kernel<<<elemBlocks4, 256, 0, stream>>>((const float4*)accA, (float4*)accB, n4);
    edge_scatter_kernel<<<scatterBlocks, 256, 0, stream>>>(erow, ecol, ew, accA, accB);
    f32_to_bf16_kernel<<<elemBlocks1, 256, 0, stream>>>(accB, hbA, nElem);

    // ---- h4 = relu(h3 @ W3 + b3)  (bf16 out) ----
    wmma_gemm_kernel<4><<<gemmBlocks256, 256, 0, stream>>>(
        hbA, w3t, b3, hbB, nullptr, GNN_N, GNN_K, GNN_K, GNN_K, 1);

    // ---- h5 = h4 @ W4 + b4 -> d_out [N x 40]  (NB=3 covers all 3 N-tiles) --
    wmma_gemm_kernel<3><<<gemmBlocksOut, 256, 0, stream>>>(
        hbB, w4t, b4, nullptr, (float*)d_out, GNN_N, GNN_CPAD, GNN_C, GNN_C, 0);

    // ---- log_softmax in place ----
    log_softmax_kernel<<<(GNN_N + 255) / 256, 256, 0, stream>>>((float*)d_out, GNN_N, GNN_C);
}
